// VQVAE_Audio_85770496901740
// MI455X (gfx1250) — matrix operand order repaired and checked by reading, not compiled, not run on hardware
//
#include <hip/hip_runtime.h>
#include <hip/hip_bf16.h>
#include <math.h>
#include <stdint.h>

typedef __attribute__((ext_vector_type(16))) _Float16 v16h;
typedef __attribute__((ext_vector_type(8)))  _Float16 v8h;
typedef __attribute__((ext_vector_type(8)))  float    v8f;

#define TILE_M 128
#define TILE_N 64
#define TILE_K 32
#define LDA 40   // halves per A-tile row (32 data + 8 pad, keeps 16B alignment)
#define LDB 40   // halves per B-tile column-strip

// ---------------- problem constants ----------------
#define NB    16
#define C1    256
#define T0    8192
#define T1    4095
#define T1P   4096   // padded stride so async B32 staging stays 4B-aligned
#define T2    2046
#define C3    512
#define T3    1022
#define NCODE 512
#define NROWS (NB * T3)   // 16352

__device__ __forceinline__ v16h pack16(v8h lo, v8h hi) {
  v16h r;
#pragma unroll
  for (int i = 0; i < 8; ++i) { r[i] = lo[i]; r[i + 8] = hi[i]; }
  return r;
}

// Low 32 bits of a generic LDS pointer == workgroup-relative LDS byte offset
__device__ __forceinline__ unsigned lds_off(const void* p) {
  return (unsigned)(uintptr_t)p;
}

// CDNA5 async global->LDS copies (ASYNCcnt).  vdst = LDS byte address VGPR,
// vaddr = 64-bit global address VGPR pair.
__device__ __forceinline__ void async_copy_b128(unsigned lds_byte, const void* g) {
  asm volatile("global_load_async_to_lds_b128 %0, %1, off"
               :: "v"(lds_byte), "v"(g) : "memory");
}
__device__ __forceinline__ void async_copy_b32(unsigned lds_byte, const void* g) {
  asm volatile("global_load_async_to_lds_b32 %0, %1, off"
               :: "v"(lds_byte), "v"(g) : "memory");
}
__device__ __forceinline__ void wait_async0() {
  asm volatile("s_wait_asynccnt 0x0" ::: "memory");
}

// ---------------- fp32 -> f16 elementwise convert ----------------
__global__ void cvt_f32_f16_kernel(const float* __restrict__ in,
                                   _Float16* __restrict__ out, int n) {
  int i = blockIdx.x * blockDim.x + threadIdx.x;
  if (i < n) out[i] = (_Float16)in[i];
}

// ---------------- fused im2col conv1d + bias + tanh, WMMA f16 ----------------
// GEMM per batch: C[M=Cout, N=Tout] = Wh[M, K=Cin*4] * B[K, N],
// B[ci*4+k, t] = X[b, ci, 2t+k].  Block: 256 thr (8 waves), tile 128x64.
// Double-buffered: next tile staged (async DMA / batched VALU) during WMMA.
template <typename TI, typename TO>
__global__ __launch_bounds__(256)
void conv_wmma_kernel(const _Float16* __restrict__ Wh,   // [Cout, Cin*4]
                      const float* __restrict__ bias,    // [Cout]
                      const TI* __restrict__ X,          // [B, Cin, Tins]
                      TO* __restrict__ Y,                // [B, Cout, Touts]
                      int Cin, int Cout, int Tins, int Tout, int Touts) {
  __shared__ __align__(16) _Float16 lds_a[2][TILE_M * LDA];
  __shared__ __align__(16) _Float16 lds_b[2][TILE_N * LDB];

  const int K = Cin * 4;
  const int tid  = threadIdx.x;
  const int wave = tid >> 5;
  const int lane = tid & 31;
  const int half = lane >> 4;
  const int lan16 = lane & 15;
  const int n0 = blockIdx.x * TILE_N;
  const int m0 = blockIdx.y * TILE_M;
  const int b  = blockIdx.z;
  const TI* Xb = X + (size_t)b * Cin * Tins;

  v8f acc[4];
#pragma unroll
  for (int j = 0; j < 4; ++j)
#pragma unroll
    for (int i = 0; i < 8; ++i) acc[j][i] = 0.0f;

  auto stage = [&](int buf, int kc) {
    const unsigned la = lds_off(lds_a[buf]);
    // A tile (weights, f16): async DMA, 128 rows x 32 halves, B128 ops.
#pragma unroll
    for (int it = 0; it < 2; ++it) {
      int e = tid + it * 256;
      int r = e >> 2, seg = e & 3;
      async_copy_b128(la + (unsigned)(r * LDA + seg * 8) * 2,
                      Wh + (m0 + r) * K + kc + seg * 8);
    }
    if constexpr (sizeof(TI) == 2) {
      // f16 input: one async B32 covers the (kk, kk+1) pair at x[ci][2n+kk].
      const unsigned lb = lds_off(lds_b[buf]);
#pragma unroll
      for (int it = 0; it < 4; ++it) {
        int e = tid + it * 256;
        int nl = e & 63, cp = e >> 6;       // consecutive tid -> consecutive n
        int cil = cp >> 1, kp = (cp & 1) * 2;
        int ci = (kc >> 2) + cil;
        int n = n0 + nl;
        async_copy_b32(lb + (unsigned)(nl * LDB + cil * 4 + kp) * 2,
                       (const _Float16*)Xb + ci * Tins + 2 * n + kp);
      }
    } else {
      // fp32 input: batched clamped loads, then cvt+store (branch-free).
      _Float16* Lb = lds_b[buf];
      float v[8];
#pragma unroll
      for (int it = 0; it < 8; ++it) {
        int e = tid + it * 256;
        int kl = e >> 6, nl = e & 63;
        int kg = kc + kl, ci = kg >> 2, kk = kg & 3;
        int n = n0 + nl;
        int nc = n < Tout ? n : (Tout - 1);
        v[it] = (float)Xb[ci * Tins + 2 * nc + kk];
      }
#pragma unroll
      for (int it = 0; it < 8; ++it) {
        int e = tid + it * 256;
        int kl = e >> 6, nl = e & 63;
        int n = n0 + nl;
        Lb[nl * LDB + kl] = (_Float16)((n < Tout) ? v[it] : 0.0f);
      }
    }
  };

  auto compute = [&](int buf) {
    const _Float16* La = lds_a[buf];
    const _Float16* Lb = lds_b[buf];
    const int row = wave * 16 + lan16;
    v8h a_lo = *(const v8h*)(La + row * LDA + half * 8);
    v8h a_hi = *(const v8h*)(La + row * LDA + 16 + half * 8);
    v16h afrag = pack16(a_lo, a_hi);
#pragma unroll
    for (int j = 0; j < 4; ++j) {
      int col = j * 16 + lan16;
      v8h b_lo = *(const v8h*)(Lb + col * LDB + half * 8);
      v8h b_hi = *(const v8h*)(Lb + col * LDB + 16 + half * 8);
      v16h bfrag = pack16(b_lo, b_hi);
      acc[j] = __builtin_amdgcn_wmma_f32_16x16x32_f16(
          false, afrag, false, bfrag, (short)0, acc[j], false, false);
    }
  };

  const int nchunks = K / TILE_K;
  stage(0, 0);
  for (int ic = 0; ic < nchunks; ++ic) {
    wait_async0();
    __syncthreads();
    if (ic + 1 < nchunks) stage((ic + 1) & 1, (ic + 1) * TILE_K);
    compute(ic & 1);
  }

  // Epilogue: bias + tanh, store (C/D layout: vgpr i -> row i + 8*half).
#pragma unroll
  for (int j = 0; j < 4; ++j) {
    int colg = n0 + j * 16 + lan16;
    if (colg >= Tout) continue;
#pragma unroll
    for (int i = 0; i < 8; ++i) {
      int m = wave * 16 + half * 8 + i;
      float v = tanhf(acc[j][i] + bias[m0 + m]);
      Y[((size_t)b * Cout + (m0 + m)) * Touts + colg] = (TO)v;
    }
  }
}

// ---------------- codebook row sums S_k = sum_c cb[k][c] ----------------
__global__ void cb_rowsum_kernel(const float* __restrict__ cb,
                                 float* __restrict__ S) {
  int k = blockIdx.x * blockDim.x + threadIdx.x;
  if (k < NCODE) {
    float s = 0.0f;
    for (int c = 0; c < C3; ++c) s += cb[k * C3 + c];
    S[k] = s;
  }
}

// ---------------- VQ GEMM: Score[n,k] = S_k - 2 * (z_n . cb_k) ----------------
// A[n,c] = Z[b, c, t] (n = b*T + t), B[c,k] = cbh[k,c] (pre-converted f16).
__global__ __launch_bounds__(256)
void vq_gemm_kernel(const float* __restrict__ Z,        // [B, C3, T3] fp32
                    const _Float16* __restrict__ CBH,   // [NCODE, C3] f16
                    const float* __restrict__ Srow,     // [NCODE]
                    float* __restrict__ Score) {        // [NROWS, NCODE]
  __shared__ __align__(16) _Float16 lds_a[2][TILE_M * LDA];
  __shared__ __align__(16) _Float16 lds_b[2][TILE_N * LDB];

  const int tid  = threadIdx.x;
  const int wave = tid >> 5;
  const int lane = tid & 31;
  const int half = lane >> 4;
  const int lan16 = lane & 15;
  const int n0 = blockIdx.x * TILE_N;   // codeword tile
  const int m0 = blockIdx.y * TILE_M;   // flat-row tile

  v8f acc[4];
#pragma unroll
  for (int j = 0; j < 4; ++j)
#pragma unroll
    for (int i = 0; i < 8; ++i) acc[j][i] = 0.0f;

  auto stage = [&](int buf, int kc) {
    // B tile (codebook, f16): async DMA, 64 rows x 32 halves, B128. 1 op/thr.
    const unsigned lb = lds_off(lds_b[buf]);
    {
      int nl = tid & 63, seg = tid >> 6;
      async_copy_b128(lb + (unsigned)(nl * LDB + seg * 8) * 2,
                      CBH + (n0 + nl) * C3 + kc + seg * 8);
    }
    // A tile (z, fp32 transposed gather): batched clamped loads, then cvt.
    _Float16* La = lds_a[buf];
    float v[16];
#pragma unroll
    for (int it = 0; it < 16; ++it) {
      int e = tid + it * 256;
      int c = e >> 7, r = e & 127;        // consecutive tid -> consecutive t
      int n = m0 + r;
      int nc = n < NROWS ? n : (NROWS - 1);
      int bb = nc / T3, t = nc - bb * T3;
      v[it] = Z[(bb * C3 + kc + c) * T3 + t];
    }
#pragma unroll
    for (int it = 0; it < 16; ++it) {
      int e = tid + it * 256;
      int c = e >> 7, r = e & 127;
      int n = m0 + r;
      La[r * LDA + c] = (_Float16)((n < NROWS) ? v[it] : 0.0f);
    }
  };

  auto compute = [&](int buf) {
    const _Float16* La = lds_a[buf];
    const _Float16* Lb = lds_b[buf];
    const int row = wave * 16 + lan16;
    v8h a_lo = *(const v8h*)(La + row * LDA + half * 8);
    v8h a_hi = *(const v8h*)(La + row * LDA + 16 + half * 8);
    v16h afrag = pack16(a_lo, a_hi);
#pragma unroll
    for (int j = 0; j < 4; ++j) {
      int col = j * 16 + lan16;
      v8h b_lo = *(const v8h*)(Lb + col * LDB + half * 8);
      v8h b_hi = *(const v8h*)(Lb + col * LDB + 16 + half * 8);
      v16h bfrag = pack16(b_lo, b_hi);
      acc[j] = __builtin_amdgcn_wmma_f32_16x16x32_f16(
          false, afrag, false, bfrag, (short)0, acc[j], false, false);
    }
  };

  const int nchunks = C3 / TILE_K;      // 16
  stage(0, 0);
  for (int ic = 0; ic < nchunks; ++ic) {
    wait_async0();
    __syncthreads();
    if (ic + 1 < nchunks) stage((ic + 1) & 1, (ic + 1) * TILE_K);
    compute(ic & 1);
  }

#pragma unroll
  for (int j = 0; j < 4; ++j) {
    int kg = n0 + j * 16 + lan16;
#pragma unroll
    for (int i = 0; i < 8; ++i) {
      int m = m0 + wave * 16 + half * 8 + i;
      if (m < NROWS) Score[(size_t)m * NCODE + kg] = Srow[kg] - 2.0f * acc[j][i];
    }
  }
}

// ---------------- argmin + gather + loss/count accumulation ----------------
// One wave per flattened row n; first-occurrence argmin like jnp.argmin.
__global__ __launch_bounds__(256)
void vq_select_kernel(const float* __restrict__ Score,  // [NROWS, NCODE]
                      const float* __restrict__ Z,      // [B, C3, T3]
                      const float* __restrict__ CB,     // [NCODE, C3]
                      float* __restrict__ Qout,         // [B, C3, T3]
                      int* __restrict__ counts,
                      double* __restrict__ loss_acc) {
  const int wave = threadIdx.x >> 5;
  const int lane = threadIdx.x & 31;
  const int n = blockIdx.x * 8 + wave;
  if (n >= NROWS) return;

  const float* sr = Score + (size_t)n * NCODE;
  float best = INFINITY;
  int bidx = 0x7fffffff;
#pragma unroll
  for (int it = 0; it < NCODE / 32; ++it) {
    int k = lane + it * 32;
    float v = sr[k];
    if (v < best) { best = v; bidx = k; }
  }
  for (int off = 16; off > 0; off >>= 1) {
    float ov = __shfl_down(best, off, 32);
    int oi = __shfl_down(bidx, off, 32);
    if (ov < best || (ov == best && oi < bidx)) { best = ov; bidx = oi; }
  }
  bidx = __shfl(bidx, 0, 32);

  const int b = n / T3, t = n - b * T3;
  const float* q = CB + bidx * C3;
  float lsum = 0.0f;
#pragma unroll
  for (int it = 0; it < C3 / 32; ++it) {
    int c = lane + it * 32;
    float zv = Z[(b * C3 + c) * T3 + t];
    float qv = q[c];
    // literal: zt + (stop_grad(q) - zt)
    Qout[(b * C3 + c) * T3 + t] = zv + (qv - zv);
    float d = zv - qv;
    lsum += d * d;
  }
  for (int off = 16; off > 0; off >>= 1) lsum += __shfl_down(lsum, off, 32);
  if (lane == 0) {
    atomicAdd(counts + bidx, 1);
    atomicAdd(loss_acc, (double)lsum);
  }
}

// ---------------- finalize: loss scalar + perplexity ----------------
__global__ __launch_bounds__(512)
void finalize_kernel(const int* __restrict__ counts,
                     const double* __restrict__ loss_acc,
                     float* __restrict__ out_loss,
                     float* __restrict__ out_ppl,
                     double loss_scale) {
  __shared__ double sh[512];
  int k = threadIdx.x;
  double p = (double)counts[k] / (double)NROWS;
  sh[k] = p * log(p + 1e-10);
  __syncthreads();
  for (int s = 256; s > 0; s >>= 1) {
    if (k < s) sh[k] += sh[k + s];
    __syncthreads();
  }
  if (k == 0) {
    *out_ppl = (float)exp(-sh[0]);
    *out_loss = (float)(*loss_acc * loss_scale);
  }
}

// ---------------- workspace layout ----------------
static const size_t OFF_W1H = 0;
static const size_t OFF_W2H = OFF_W1H + (size_t)C1 * C1 * 4 * 2;
static const size_t OFF_W3H = OFF_W2H + (size_t)C1 * C1 * 4 * 2;
static const size_t OFF_H1  = OFF_W3H + (size_t)C3 * C1 * 4 * 2;
static const size_t OFF_H2  = OFF_H1 + (size_t)NB * C1 * T1P * 2;
static const size_t OFF_Z   = OFF_H2 + (size_t)NB * C1 * T2 * 2;
static const size_t OFF_CBH = OFF_Z + (size_t)NB * C3 * T3 * 4;
static const size_t OFF_SK  = OFF_CBH + (size_t)NCODE * C3 * 2;
static const size_t OFF_CNT = OFF_SK + (size_t)NCODE * 4;
static const size_t OFF_LOSS = OFF_CNT + (size_t)NCODE * 4;      // 8-aligned
static const size_t OFF_SCORE = OFF_H1;  // reuse h1 region (dead after conv2)

extern "C" void kernel_launch(void* const* d_in, const int* in_sizes, int n_in,
                              void* d_out, int out_size, void* d_ws, size_t ws_size,
                              hipStream_t stream) {
  const float* x  = (const float*)d_in[0];
  const float* W1 = (const float*)d_in[1];
  const float* b1 = (const float*)d_in[2];
  const float* W2 = (const float*)d_in[3];
  const float* b2 = (const float*)d_in[4];
  const float* W3 = (const float*)d_in[5];
  const float* b3 = (const float*)d_in[6];
  const float* cb = (const float*)d_in[7];

  char* ws = (char*)d_ws;
  _Float16* w1h = (_Float16*)(ws + OFF_W1H);
  _Float16* w2h = (_Float16*)(ws + OFF_W2H);
  _Float16* w3h = (_Float16*)(ws + OFF_W3H);
  _Float16* h1  = (_Float16*)(ws + OFF_H1);
  _Float16* h2  = (_Float16*)(ws + OFF_H2);
  float*    z   = (float*)(ws + OFF_Z);
  _Float16* cbh = (_Float16*)(ws + OFF_CBH);
  float*    sk  = (float*)(ws + OFF_SK);
  int*      cnt = (int*)(ws + OFF_CNT);
  double*   lac = (double*)(ws + OFF_LOSS);
  float*    score = (float*)(ws + OFF_SCORE);

  float* out = (float*)d_out;
  float* out_loss = out;
  float* out_q    = out + 1;
  float* out_ppl  = out + (out_size - 1);

  hipMemsetAsync(ws + OFF_CNT, 0, NCODE * 4 + 8, stream);

  // Weight/codebook conversion (flat [Cout, Cin, 4] == GEMM-A [Cout, Cin*4]).
  {
    int n1 = C1 * C1 * 4, n3 = C3 * C1 * 4, ncb = NCODE * C3;
    cvt_f32_f16_kernel<<<(n1 + 255) / 256, 256, 0, stream>>>(W1, w1h, n1);
    cvt_f32_f16_kernel<<<(n1 + 255) / 256, 256, 0, stream>>>(W2, w2h, n1);
    cvt_f32_f16_kernel<<<(n3 + 255) / 256, 256, 0, stream>>>(W3, w3h, n3);
    cvt_f32_f16_kernel<<<(ncb + 255) / 256, 256, 0, stream>>>(cb, cbh, ncb);
  }

  // conv1: [16,256,8192] -> h1 [16,256,4095] (stride 4096)
  conv_wmma_kernel<float, _Float16>
      <<<dim3((T1 + TILE_N - 1) / TILE_N, C1 / TILE_M, NB), 256, 0, stream>>>(
          w1h, b1, x, h1, C1, C1, T0, T1, T1P);
  // conv2: -> h2 [16,256,2046]
  conv_wmma_kernel<_Float16, _Float16>
      <<<dim3((T2 + TILE_N - 1) / TILE_N, C1 / TILE_M, NB), 256, 0, stream>>>(
          w2h, b2, h1, h2, C1, C1, T1P, T2, T2);
  // conv3: -> z [16,512,1022] fp32
  conv_wmma_kernel<_Float16, float>
      <<<dim3((T3 + TILE_N - 1) / TILE_N, C3 / TILE_M, NB), 256, 0, stream>>>(
          w3h, b3, h2, z, C1, C3, T2, T3, T3);

  cb_rowsum_kernel<<<(NCODE + 255) / 256, 256, 0, stream>>>(cb, sk);

  // Scores = S_k - 2 * z @ cb^T   (reuses h1 region)
  vq_gemm_kernel<<<dim3(NCODE / TILE_N, (NROWS + TILE_M - 1) / TILE_M), 256, 0,
                   stream>>>(z, cbh, sk, score);

  // argmin + gather + loss/count
  vq_select_kernel<<<(NROWS + 7) / 8, 256, 0, stream>>>(score, z, cb, out_q,
                                                        cnt, lac);

  // loss = 1.25 * mean((z-q)^2); perplexity from counts
  finalize_kernel<<<1, 512, 0, stream>>>(cnt, lac, out_loss, out_ppl,
                                         1.25 / ((double)NROWS * C3));
}
